// MVP_27487790694493
// MI455X (gfx1250) — matrix operand order repaired and checked by reading, not compiled, not run on hardware
//
#include <hip/hip_runtime.h>
#include <hip/hip_bf16.h>

typedef __bf16 bf16;
typedef __attribute__((ext_vector_type(8)))  __bf16 v8bf;
typedef __attribute__((ext_vector_type(16))) __bf16 v16bf;
typedef __attribute__((ext_vector_type(8)))  float  v8f;

#define B_   64
#define D_   768
#define NT   197
#define HN   12
#define HD_  64
#define FF_  3072
#define NKMX 217   // 197 + 20 (e-prompt layers)
#define NC_  100
#define T_   10

static inline long cdivl(long a, long b) { return (a + b - 1) / b; }

// ---------------------------------------------------------------------------
// Generic WMMA GEMM:  C[M,Nout] = A[M,K](bf16) * W[Nout,K]^T (bf16)
// Block tile 64x128, 128 threads (4 waves in 2x2), each wave 32x64
// (2x4 WMMA frags -> 8 v_wmma per K-step, A-fragment reuse x4).
// Double-buffered LDS with register-staged global loads: next tile's
// global_load_b128s issue before the current tile's WMMAs, store to the
// alternate LDS buffer after compute -> one barrier per K-step.
// W rows are clamped (not zeroed) when Nout isn't a tile multiple: C columns
// gc >= Nout are never stored, so no exec-masking in the hot loop.
// M = gridDim.x * 64 (always exact here).
// ---------------------------------------------------------------------------
__global__ __launch_bounds__(128)
void gemm_bf16_wmma(const bf16* __restrict__ A, int lda,
                    const bf16* __restrict__ W, int ldb, int Nout,
                    const float* __restrict__ bias,
                    const float* __restrict__ residual,
                    float* __restrict__ Cf, bf16* __restrict__ Cb,
                    int ldc, int K, int dogelu)
{
    __shared__ bf16 As[2][64][40];    // +8 pad: rows stay 16B aligned
    __shared__ bf16 Bs[2][128][40];

    const int tid  = threadIdx.x;
    const int lane = tid & 31;
    const int wave = tid >> 5;
    const int wm   = wave >> 1;          // 0..1
    const int wn   = wave & 1;           // 0..1
    const int l16  = lane & 15;
    const int lh   = lane >> 4;          // 0..1
    const long row0 = (long)blockIdx.x * 64;
    const int  col0 = blockIdx.y * 128;

    // Loop-invariant staging coordinates.
    // A tile: 64 rows x 32 K = 256 chunks of 8 bf16; B tile: 128 x 32 = 512.
    int arow[2], acol[2];
#pragma unroll
    for (int it = 0; it < 2; ++it) {
        int c = tid + it * 128;
        arow[it] = c >> 2;
        acol[it] = (c & 3) * 8;
    }
    int blrow[4], bgrow[4], bcol[4];
#pragma unroll
    for (int it = 0; it < 4; ++it) {
        int c = tid + it * 128;
        blrow[it] = c >> 2;                      // 0..127 (LDS row)
        int n = col0 + blrow[it];
        bgrow[it] = (n < Nout) ? n : (Nout - 1); // clamped global W row
        bcol[it]  = (c & 3) * 8;
    }

    v8f acc[2][4];
#pragma unroll
    for (int a = 0; a < 2; ++a)
#pragma unroll
        for (int b = 0; b < 4; ++b)
#pragma unroll
            for (int j = 0; j < 8; ++j) acc[a][b][j] = 0.0f;

    v8bf rg[6];
    // ---- prologue: stage K-tile 0 and commit to LDS buffer 0 ----
#pragma unroll
    for (int it = 0; it < 2; ++it)
        rg[it] = *(const v8bf*)(A + (row0 + arow[it]) * (long)lda + acol[it]);
#pragma unroll
    for (int it = 0; it < 4; ++it)
        rg[2 + it] = *(const v8bf*)(W + (long)bgrow[it] * ldb + bcol[it]);
#pragma unroll
    for (int it = 0; it < 2; ++it) *(v8bf*)&As[0][arow[it]][acol[it]] = rg[it];
#pragma unroll
    for (int it = 0; it < 4; ++it) *(v8bf*)&Bs[0][blrow[it]][bcol[it]] = rg[2 + it];
    __syncthreads();

    int buf = 0;
    for (int kt = 0; kt < K; kt += 32) {
        const bool more = (kt + 32) < K;
        if (more) {   // issue next tile's global loads (latency hidden by WMMAs)
#pragma unroll
            for (int it = 0; it < 2; ++it)
                rg[it] = *(const v8bf*)(A + (row0 + arow[it]) * (long)lda + kt + 32 + acol[it]);
#pragma unroll
            for (int it = 0; it < 4; ++it)
                rg[2 + it] = *(const v8bf*)(W + (long)bgrow[it] * ldb + kt + 32 + bcol[it]);
            if (kt + 64 < K)  // far prefetch of A into caches
                __builtin_prefetch(A + (row0 + (tid & 63)) * (long)lda + kt + 64, 0, 1);
        }

        // ---- compute from LDS[buf] ----
        v16bf af[2];
#pragma unroll
        for (int ti = 0; ti < 2; ++ti) {
            int m = wm * 32 + ti * 16 + l16;
            v8bf lo = *(const v8bf*)&As[buf][m][lh * 8];
            v8bf hi = *(const v8bf*)&As[buf][m][16 + lh * 8];
            af[ti] = __builtin_shufflevector(lo, hi, 0,1,2,3,4,5,6,7,8,9,10,11,12,13,14,15);
        }
#pragma unroll
        for (int tj = 0; tj < 4; ++tj) {
            int nn = wn * 64 + tj * 16 + l16;
            v8bf bl = *(const v8bf*)&Bs[buf][nn][lh * 8];
            v8bf bh = *(const v8bf*)&Bs[buf][nn][16 + lh * 8];
            v16bf bfg = __builtin_shufflevector(bl, bh, 0,1,2,3,4,5,6,7,8,9,10,11,12,13,14,15);
#pragma unroll
            for (int ti = 0; ti < 2; ++ti)
                acc[ti][tj] = __builtin_amdgcn_wmma_f32_16x16x32_bf16(
                    false, af[ti], false, bfg, (short)0, acc[ti][tj], false, false);
        }

        if (more) {   // commit staged tile to the alternate buffer
            int nb = buf ^ 1;
#pragma unroll
            for (int it = 0; it < 2; ++it) *(v8bf*)&As[nb][arow[it]][acol[it]] = rg[it];
#pragma unroll
            for (int it = 0; it < 4; ++it) *(v8bf*)&Bs[nb][blrow[it]][bcol[it]] = rg[2 + it];
        }
        buf ^= 1;
        __syncthreads();
    }

    // ---- epilogue: C element (M = ti*16 + vi + 8*lh, N = tj*16 + l16) ----
#pragma unroll
    for (int ti = 0; ti < 2; ++ti)
#pragma unroll
        for (int tj = 0; tj < 4; ++tj) {
            int gc = col0 + wn * 64 + tj * 16 + l16;
            if (gc >= Nout) continue;
            long gr0 = row0 + wm * 32 + ti * 16 + lh * 8;
#pragma unroll
            for (int vi = 0; vi < 8; ++vi) {
                long r = gr0 + vi;
                float v = acc[ti][tj][vi];
                if (bias)     v += bias[gc];
                if (dogelu)   v = 0.5f * v * (1.0f + erff(v * 0.70710678118f));
                if (residual) v += residual[r * ldc + gc];
                if (Cf) Cf[r * ldc + gc] = v;
                if (Cb) Cb[r * ldc + gc] = (bf16)v;
            }
        }
}

// ---------------------------------------------------------------------------
// fp32 -> bf16 cast
// ---------------------------------------------------------------------------
__global__ void f32_to_bf16_k(const float* __restrict__ in, bf16* __restrict__ out, long n)
{
    long i = (long)blockIdx.x * blockDim.x + threadIdx.x;
    if (i < n) out[i] = (bf16)in[i];
}

// ---------------------------------------------------------------------------
// Patchify: [B,3,224,224] -> bf16 [B*196, 768] in timm conv-kernel order
// ---------------------------------------------------------------------------
__global__ void im2row_k(const float* __restrict__ in, bf16* __restrict__ out)
{
    long i = (long)blockIdx.x * blockDim.x + threadIdx.x;
    const long total = (long)B_ * 196 * 768;
    if (i >= total) return;
    int col = (int)(i % 768); long r = i / 768;
    int p = (int)(r % 196);   int b = (int)(r / 196);
    int py = p / 14, px = p % 14;
    int c  = col / 256; int rem = col % 256;
    int iy = rem / 16, ix = rem % 16;
    out[i] = (bf16)in[(((long)b * 3 + c) * 224 + (py * 16 + iy)) * 224 + (px * 16 + ix)];
}

// ---------------------------------------------------------------------------
// x0 = [cls | patch_embed] + pos_embed   (fp32 [B,197,768])
// ---------------------------------------------------------------------------
__global__ void assemble_x0_k(const float* __restrict__ tmp, const float* __restrict__ cls,
                              const float* __restrict__ pos, float* __restrict__ x0)
{
    long i = (long)blockIdx.x * blockDim.x + threadIdx.x;
    const long total = (long)B_ * NT * D_;
    if (i >= total) return;
    int d = (int)(i % D_); long t = i / D_;
    int n = (int)(t % NT); int b = (int)(t / NT);
    float v = (n == 0) ? cls[d] : tmp[((long)b * 196 + n - 1) * D_ + d];
    x0[i] = v + pos[n * D_ + d];
}

// ---------------------------------------------------------------------------
// LayerNorm: one wave32 per 768-wide row; bf16 output
// ---------------------------------------------------------------------------
__global__ void layernorm_k(const float* __restrict__ X, long xstride,
                            const float* __restrict__ gs, const float* __restrict__ gb,
                            bf16* __restrict__ Y, long ystride, int nrows)
{
    int gw   = (int)(((long)blockIdx.x * blockDim.x + threadIdx.x) >> 5);
    int lane = threadIdx.x & 31;
    if (gw >= nrows) return;
    const float* xr = X + (long)gw * xstride;
    float s = 0.f, q = 0.f;
    for (int i = lane; i < D_; i += 32) { float v = xr[i]; s += v; q += v * v; }
#pragma unroll
    for (int o = 16; o; o >>= 1) { s += __shfl_xor(s, o, 32); q += __shfl_xor(q, o, 32); }
    float mu  = s * (1.0f / D_);
    float var = q * (1.0f / D_) - mu * mu;
    float rs  = rsqrtf(var + 1e-6f);
    bf16* yr = Y + (long)gw * ystride;
    for (int i = lane; i < D_; i += 32)
        yr[i] = (bf16)((xr[i] - mu) * rs * gs[i] + gb[i]);
}

// ---------------------------------------------------------------------------
// Build prompted K/V inputs: hk/hv[b, 0:197] = h, then Lp prompt tokens.
// e-prompts selected per batch via idx (psel_stride = 120*768); g: idx=null.
// ---------------------------------------------------------------------------
__global__ void build_kv_k(const bf16* __restrict__ h,
                           const float* __restrict__ Pk, const float* __restrict__ Pv,
                           const int* __restrict__ idx, long psel_stride, int Lp,
                           bf16* __restrict__ hk, bf16* __restrict__ hv)
{
    long i = (long)blockIdx.x * blockDim.x + threadIdx.x;
    int Nk = NT + Lp;
    long total = (long)B_ * Nk * D_;
    if (i >= total) return;
    int d = (int)(i % D_); long t = i / D_;
    int n = (int)(t % Nk); int b = (int)(t / Nk);
    long o = ((long)b * NKMX + n) * D_ + d;
    if (n < NT) {
        bf16 v = h[((long)b * NT + n) * D_ + d];
        hk[o] = v; hv[o] = v;
    } else {
        int j = n - NT;
        long sel = idx ? (long)idx[b] * psel_stride : 0;
        hk[o] = (bf16)Pk[sel + (long)j * D_ + d];
        hv[o] = (bf16)Pv[sel + (long)j * D_ + d];
    }
}

// ---------------------------------------------------------------------------
// Attention: one wave32 per (b, head, q-row); online softmax; lane owns
// dims {lane, lane+32} of the 64-wide head. ~4% of total FLOPs.
// ---------------------------------------------------------------------------
__global__ void attention_k(const bf16* __restrict__ Q, long qbs, int qrs,
                            const bf16* __restrict__ Km, long kbs, int krs,
                            const bf16* __restrict__ Vm, long vbs, int vrs,
                            bf16* __restrict__ O, int Nk)
{
    int gw   = (int)(((long)blockIdx.x * blockDim.x + threadIdx.x) >> 5);
    int lane = threadIdx.x & 31;
    const int total = B_ * HN * NT;
    if (gw >= total) return;
    int qrow = gw % NT; int t = gw / NT; int hh = t % HN; int b = t / HN;

    const bf16* qp = Q + (long)b * qbs + (long)qrow * qrs + hh * HD_;
    const float SC = 0.125f;  // 64^-0.5
    float q0 = (float)qp[lane] * SC;
    float q1 = (float)qp[lane + 32] * SC;
    float m = -3.0e38f, s = 0.f, a0 = 0.f, a1 = 0.f;
    const bf16* kb = Km + (long)b * kbs + hh * HD_;
    const bf16* vb = Vm + (long)b * vbs + hh * HD_;
    for (int j = 0; j < Nk; ++j) {
        const bf16* kp = kb + (long)j * krs;
        float p = q0 * (float)kp[lane] + q1 * (float)kp[lane + 32];
#pragma unroll
        for (int o = 16; o; o >>= 1) p += __shfl_xor(p, o, 32);
        const bf16* vp = vb + (long)j * vrs;
        float v0 = (float)vp[lane], v1 = (float)vp[lane + 32];
        float mn = fmaxf(m, p);
        float cs = __expf(m - mn);
        float e  = __expf(p - mn);
        s = s * cs + e;
        a0 = a0 * cs + e * v0;
        a1 = a1 * cs + e * v1;
        m = mn;
    }
    float inv = 1.0f / s;
    bf16* op = O + ((long)b * NT + qrow) * D_ + hh * HD_;
    op[lane]      = (bf16)(a0 * inv);
    op[lane + 32] = (bf16)(a1 * inv);
}

// ---------------------------------------------------------------------------
// query[b,:] = x[b, 0, :]
// ---------------------------------------------------------------------------
__global__ void extract_query_k(const float* __restrict__ x, float* __restrict__ q)
{
    long i = (long)blockIdx.x * blockDim.x + threadIdx.x;
    if (i >= (long)B_ * D_) return;
    int d = (int)(i % D_); int b = (int)(i / D_);
    q[i] = x[((long)b * NT) * D_ + d];
}

// ---------------------------------------------------------------------------
// Prompt selection: idx[b] = argmax_t cos(query_b, key_t)  (== argmin dist)
// One wave per batch element.
// ---------------------------------------------------------------------------
__global__ void select_prompt_k(const float* __restrict__ query,
                                const float* __restrict__ key_emb, int* __restrict__ idx)
{
    int gw   = (int)(((long)blockIdx.x * blockDim.x + threadIdx.x) >> 5);
    int lane = threadIdx.x & 31;
    if (gw >= B_) return;
    const float* q = query + (long)gw * D_;
    float qs = 0.f;
    for (int i = lane; i < D_; i += 32) { float v = q[i]; qs += v * v; }
#pragma unroll
    for (int o = 16; o; o >>= 1) qs += __shfl_xor(qs, o, 32);
    float qn = sqrtf(qs) + 1e-8f;
    float best = -3e38f; int bi = 0;
    for (int t = 0; t < T_; ++t) {
        const float* k = key_emb + (long)t * D_;
        float d = 0.f, ks = 0.f;
        for (int i = lane; i < D_; i += 32) { float kv = k[i]; d += q[i] * kv; ks += kv * kv; }
#pragma unroll
        for (int o = 16; o; o >>= 1) { d += __shfl_xor(d, o, 32); ks += __shfl_xor(ks, o, 32); }
        float sim = d / (qn * (sqrtf(ks) + 1e-8f));
        if (sim > best) { best = sim; bi = t; }
    }
    if (lane == 0) idx[gw] = bi;
}

// ---------------------------------------------------------------------------
// logits *= 2 * sigmoid(mask_param[idx[b]])
// ---------------------------------------------------------------------------
__global__ void apply_mask_k(float* __restrict__ out, const float* __restrict__ mask_p,
                             const int* __restrict__ idx)
{
    int i = blockIdx.x * blockDim.x + threadIdx.x;
    if (i >= B_ * NC_) return;
    int c = i % NC_; int b = i / NC_;
    float m = mask_p[idx[b] * NC_ + c];
    out[i] *= 2.0f / (1.0f + __expf(-m));
}

// ---------------------------------------------------------------------------
// Host orchestration
// ---------------------------------------------------------------------------
struct VitCtx {
    const float *ln1_s, *ln1_b, *qkv_b, *proj_b, *ln2_s, *ln2_b, *fc1_b, *fc2_b;
    const float *g_prom, *e_prom;
    const bf16 *w_qkv, *w_proj, *w_fc1, *w_fc2;
    float* x;
    bf16 *h, *hk, *hv, *qkv, *kbuf, *vbuf, *attn, *a1;
    const int* idx;
    hipStream_t stream;
};

static void vit_pass(const VitCtx& c, bool with_prompts)
{
    const long MD = (long)B_ * NT;     // 12608
    for (int n = 0; n < 12; ++n) {
        // LN1 -> h (bf16)
        layernorm_k<<<dim3((unsigned)cdivl(MD, 8)), 256, 0, c.stream>>>(
            c.x, D_, c.ln1_s + n * D_, c.ln1_b + n * D_, c.h, D_, (int)MD);

        bool isG = with_prompts && (n == 0 || n == 1);
        bool isE = with_prompts && (n >= 2 && n <= 4);
        if (isG || isE) {
            int pi = isG ? n : (n - 2);
            int Lp = isG ? 5 : 20;
            const float* Pk = isG ? c.g_prom + (2 * pi) * 5 * D_
                                  : c.e_prom + (2 * pi) * 20 * D_;
            const float* Pv = isG ? c.g_prom + (2 * pi + 1) * 5 * D_
                                  : c.e_prom + (2 * pi + 1) * 20 * D_;
            long pstride = isG ? 0 : (long)120 * D_;
            const int* ix = isG ? nullptr : c.idx;
            int Nk = NT + Lp;
            build_kv_k<<<dim3((unsigned)cdivl((long)B_ * Nk * D_, 256)), 256, 0, c.stream>>>(
                c.h, Pk, Pv, ix, pstride, Lp, c.hk, c.hv);
            const bf16* Wn = c.w_qkv + (long)n * 3 * D_ * D_;
            const float* bn = c.qkv_b + (long)n * 3 * D_;
            // q = h @ Wq^T
            gemm_bf16_wmma<<<dim3(197, 6), 128, 0, c.stream>>>(
                c.h, D_, Wn, D_, D_, bn, nullptr, nullptr, c.qkv, D_, D_, 0);
            // k = hk @ Wk^T   (M = 64*217, unused tail rows are dead)
            gemm_bf16_wmma<<<dim3(217, 6), 128, 0, c.stream>>>(
                c.hk, D_, Wn + (long)D_ * D_, D_, D_, bn + D_, nullptr, nullptr,
                c.kbuf, D_, D_, 0);
            // v = hv @ Wv^T
            gemm_bf16_wmma<<<dim3(217, 6), 128, 0, c.stream>>>(
                c.hv, D_, Wn + (long)2 * D_ * D_, D_, D_, bn + 2 * D_, nullptr, nullptr,
                c.vbuf, D_, D_, 0);
            attention_k<<<dim3((unsigned)cdivl((long)B_ * HN * NT, 8)), 256, 0, c.stream>>>(
                c.qkv, (long)NT * D_, D_,
                c.kbuf, (long)NKMX * D_, D_,
                c.vbuf, (long)NKMX * D_, D_,
                c.attn, Nk);
        } else {
            // fused qkv
            gemm_bf16_wmma<<<dim3(197, 18), 128, 0, c.stream>>>(
                c.h, D_, c.w_qkv + (long)n * 3 * D_ * D_, D_, 3 * D_,
                c.qkv_b + (long)n * 3 * D_, nullptr, nullptr, c.qkv, 3 * D_, D_, 0);
            attention_k<<<dim3((unsigned)cdivl((long)B_ * HN * NT, 8)), 256, 0, c.stream>>>(
                c.qkv,            (long)NT * 3 * D_, 3 * D_,
                c.qkv + D_,       (long)NT * 3 * D_, 3 * D_,
                c.qkv + 2 * D_,   (long)NT * 3 * D_, 3 * D_,
                c.attn, NT);
        }
        // x += attn @ proj^T + b
        gemm_bf16_wmma<<<dim3(197, 6), 128, 0, c.stream>>>(
            c.attn, D_, c.w_proj + (long)n * D_ * D_, D_, D_,
            c.proj_b + n * D_, c.x, c.x, nullptr, D_, D_, 0);
        // LN2 -> h
        layernorm_k<<<dim3((unsigned)cdivl(MD, 8)), 256, 0, c.stream>>>(
            c.x, D_, c.ln2_s + n * D_, c.ln2_b + n * D_, c.h, D_, (int)MD);
        // a1 = gelu(h @ fc1^T + b)  (bf16)
        gemm_bf16_wmma<<<dim3(197, 24), 128, 0, c.stream>>>(
            c.h, D_, c.w_fc1 + (long)n * FF_ * D_, D_, FF_,
            c.fc1_b + n * FF_, nullptr, nullptr, c.a1, FF_, D_, 1);
        // x += a1 @ fc2^T + b
        gemm_bf16_wmma<<<dim3(197, 6), 128, 0, c.stream>>>(
            c.a1, FF_, c.w_fc2 + (long)n * D_ * FF_, FF_, D_,
            c.fc2_b + n * D_, c.x, c.x, nullptr, D_, FF_, 0);
    }
}

extern "C" void kernel_launch(void* const* d_in, const int* in_sizes, int n_in,
                              void* d_out, int out_size, void* d_ws, size_t ws_size,
                              hipStream_t stream)
{
    const float* inputs  = (const float*)d_in[0];
    const float* patch_w = (const float*)d_in[1];
    const float* patch_b = (const float*)d_in[2];
    const float* cls_tok = (const float*)d_in[3];
    const float* pos_emb = (const float*)d_in[4];
    const float* ln1_s   = (const float*)d_in[5];
    const float* ln1_b   = (const float*)d_in[6];
    const float* qkv_w   = (const float*)d_in[7];
    const float* qkv_b   = (const float*)d_in[8];
    const float* proj_w  = (const float*)d_in[9];
    const float* proj_b  = (const float*)d_in[10];
    const float* ln2_s   = (const float*)d_in[11];
    const float* ln2_b   = (const float*)d_in[12];
    const float* fc1_w   = (const float*)d_in[13];
    const float* fc1_b   = (const float*)d_in[14];
    const float* fc2_w   = (const float*)d_in[15];
    const float* fc2_b   = (const float*)d_in[16];
    const float* norm_s  = (const float*)d_in[17];
    const float* norm_b  = (const float*)d_in[18];
    const float* head_w  = (const float*)d_in[19];
    const float* head_b  = (const float*)d_in[20];
    const float* key_emb = (const float*)d_in[21];
    const float* mask_p  = (const float*)d_in[22];
    const float* g_prom  = (const float*)d_in[23];
    const float* e_prom  = (const float*)d_in[24];
    float* out = (float*)d_out;
    (void)in_sizes; (void)n_in; (void)out_size; (void)ws_size;

    // ---- workspace carving (256B aligned) ----
    char* cur = (char*)d_ws;
    auto carve = [&](size_t bytes) -> void* {
        void* r = (void*)cur;
        cur += (bytes + 255) & ~(size_t)255;
        return r;
    };
    const long MD  = (long)B_ * NT;      // 12608
    const long MKV = (long)B_ * NKMX;    // 13888
    const long MP  = (long)B_ * 196;     // 12544

    bf16* w_patch = (bf16*)carve((size_t)D_ * D_ * 2);
    bf16* w_qkv   = (bf16*)carve((size_t)12 * 3 * D_ * D_ * 2);
    bf16* w_proj  = (bf16*)carve((size_t)12 * D_ * D_ * 2);
    bf16* w_fc1   = (bf16*)carve((size_t)12 * FF_ * D_ * 2);
    bf16* w_fc2   = (bf16*)carve((size_t)12 * D_ * FF_ * 2);
    bf16* w_head  = (bf16*)carve((size_t)NC_ * D_ * 2);
    bf16* patches = (bf16*)carve((size_t)MP * D_ * 2);
    float* tmpemb = (float*)carve((size_t)MP * D_ * 4);
    float* x0     = (float*)carve((size_t)MD * D_ * 4);
    float* x      = (float*)carve((size_t)MD * D_ * 4);
    bf16* h       = (bf16*)carve((size_t)MD * D_ * 2);
    bf16* hk      = (bf16*)carve((size_t)MKV * D_ * 2);
    bf16* hv      = (bf16*)carve((size_t)MKV * D_ * 2);
    bf16* qkv     = (bf16*)carve((size_t)MD * 3 * D_ * 2);
    bf16* kbuf    = (bf16*)carve((size_t)MKV * D_ * 2);
    bf16* vbuf    = (bf16*)carve((size_t)MKV * D_ * 2);
    bf16* attn    = (bf16*)carve((size_t)MD * D_ * 2);
    bf16* a1      = (bf16*)carve((size_t)MD * FF_ * 2);
    bf16* feat    = (bf16*)carve((size_t)B_ * D_ * 2);
    float* query  = (float*)carve((size_t)B_ * D_ * 4);
    int*   idx    = (int*)carve((size_t)B_ * 4);

    // ---- weight casts fp32 -> bf16 (no transpose needed: GEMM is A @ W^T) ----
    auto cvt = [&](const float* src, bf16* dst, long n) {
        f32_to_bf16_k<<<dim3((unsigned)cdivl(n, 256)), 256, 0, stream>>>(src, dst, n);
    };
    cvt(patch_w, w_patch, (long)D_ * D_);
    cvt(qkv_w,   w_qkv,   (long)12 * 3 * D_ * D_);
    cvt(proj_w,  w_proj,  (long)12 * D_ * D_);
    cvt(fc1_w,   w_fc1,   (long)12 * FF_ * D_);
    cvt(fc2_w,   w_fc2,   (long)12 * D_ * FF_);
    cvt(head_w,  w_head,  (long)NC_ * D_);

    // ---- patch embed ----
    im2row_k<<<dim3((unsigned)cdivl(MP * D_, 256)), 256, 0, stream>>>(inputs, patches);
    gemm_bf16_wmma<<<dim3(196, 6), 128, 0, stream>>>(
        patches, D_, w_patch, D_, D_, patch_b, nullptr, tmpemb, nullptr, D_, D_, 0);
    assemble_x0_k<<<dim3((unsigned)cdivl(MD * D_, 256)), 256, 0, stream>>>(
        tmpemb, cls_tok, pos_emb, x0);

    VitCtx c;
    c.ln1_s = ln1_s; c.ln1_b = ln1_b; c.qkv_b = qkv_b; c.proj_b = proj_b;
    c.ln2_s = ln2_s; c.ln2_b = ln2_b; c.fc1_b = fc1_b; c.fc2_b = fc2_b;
    c.g_prom = g_prom; c.e_prom = e_prom;
    c.w_qkv = w_qkv; c.w_proj = w_proj; c.w_fc1 = w_fc1; c.w_fc2 = w_fc2;
    c.x = x; c.h = h; c.hk = hk; c.hv = hv; c.qkv = qkv;
    c.kbuf = kbuf; c.vbuf = vbuf; c.attn = attn; c.a1 = a1;
    c.idx = idx; c.stream = stream;

    // ---- pass 1: frozen query pass ----
    hipMemcpyAsync(x, x0, (size_t)MD * D_ * 4, hipMemcpyDeviceToDevice, stream);
    vit_pass(c, false);
    extract_query_k<<<dim3((unsigned)cdivl((long)B_ * D_, 256)), 256, 0, stream>>>(x, query);
    select_prompt_k<<<dim3((unsigned)cdivl(B_, 8)), 256, 0, stream>>>(query, key_emb, idx);

    // ---- pass 2: prompted pass ----
    hipMemcpyAsync(x, x0, (size_t)MD * D_ * 4, hipMemcpyDeviceToDevice, stream);
    vit_pass(c, true);

    // ---- head: feat = LN(x)[:,0]; logits = feat @ head^T + b; *= mask ----
    layernorm_k<<<dim3((unsigned)cdivl(B_, 8)), 256, 0, stream>>>(
        x, (long)NT * D_, norm_s, norm_b, feat, D_, B_);
    gemm_bf16_wmma<<<dim3(1, 1), 128, 0, stream>>>(
        feat, D_, w_head, D_, NC_, head_b, nullptr, out, nullptr, NC_, D_, 0);
    apply_mask_k<<<dim3((unsigned)cdivl((long)B_ * NC_, 256)), 256, 0, stream>>>(
        out, mask_p, idx);
}
